// EnsembleRSSM_29686813950192
// MI455X (gfx1250) — matrix operand order repaired and checked by reading, not compiled, not run on hardware
//
#include <hip/hip_runtime.h>
#include <hip/hip_bf16.h>

// ---------------------------------------------------------------------------
// EnsembleRSSM (DreamerV2) for MI455X / gfx1250.
//   * Prologue: convert+transpose weights fp32 [K,N] -> bf16 [N,K] (L2-resident).
//   * GEMMs: v_wmma_f32_16x16x32_bf16, 32x32 tile per wave (4 WMMAs/K-step),
//     8 waves -> 64x128 block tile.
//   * Big-K GEMMs: A tile staged in LDS via global_load_async_to_lds_b128
//     (double-buffered, ASYNCcnt), B streamed from pre-transposed bf16 weights.
//   * Fused LN+ELU / GRU / softplus-head epilogues; scan ordering via stream.
// ---------------------------------------------------------------------------

typedef __bf16 bf16_t;
typedef __attribute__((ext_vector_type(16))) __bf16 v16bf;
typedef __attribute__((ext_vector_type(8)))  __bf16 v8bf;
typedef __attribute__((ext_vector_type(8)))  float  v8f;

#define TDIM 64
#define BDIM 256
#define ODIM 1024
#define ADIM 32
#define DDIM 1536
#define HDIM 1536
#define SDIM 64
#define EDIM 5
#define OUTC (4 * SDIM + DDIM) /* 1792 */

#define LDS_AROW 40 /* padded LDS row stride in bf16 (32 data + 8 pad) */

__device__ __forceinline__ bf16_t f2bf(float f) {
  unsigned u = __builtin_bit_cast(unsigned, f);
  unsigned r = u + 0x7FFFu + ((u >> 16) & 1u); // round-to-nearest-even
  unsigned short h = (unsigned short)(r >> 16);
  return __builtin_bit_cast(bf16_t, h);
}

// A fragment (16x32 bf16, CDNA5 ISA 7.12.2): elements 0-7 at p[0..7],
// elements 8-15 at p[16..23] (per-lane base already includes row & hi-half).
__device__ __forceinline__ v16bf load_afrag(const bf16_t* p) {
  v8bf lo = *(const v8bf*)(p);
  v8bf hi = *(const v8bf*)(p + 16);
  return __builtin_shufflevector(lo, hi, 0, 1, 2, 3, 4, 5, 6, 7, 8, 9, 10, 11,
                                 12, 13, 14, 15);
}

__device__ __forceinline__ v8f wmma_bf16(v16bf a, v16bf b, v8f c) {
  return __builtin_amdgcn_wmma_f32_16x16x32_bf16(false, a, false, b, (short)0,
                                                 c, false, false);
}

// C/D layout: VGPR r -> row r + 8*(lane/16); col = lane%16.
__device__ __forceinline__ void store_cfrag(float* __restrict__ C, int ldc,
                                            int row0, int col0, int lane,
                                            v8f acc) {
  int chi = (lane >> 4) * 8;
  int col = col0 + (lane & 15);
#pragma unroll
  for (int r = 0; r < 8; ++r) {
    C[(long)(row0 + chi + r) * ldc + col] = acc[r];
  }
}

// ---------------------------------------------------------------------------
// Weight convert + transpose: W fp32 [K,N] -> Wt bf16 [N,K]; batched (grid.z).
// ---------------------------------------------------------------------------
__global__ void k_w2bft(const float* __restrict__ W, bf16_t* __restrict__ Wt,
                        int K, int N) {
  long total = (long)K * N;
  long idx = (long)blockIdx.x * 256 + threadIdx.x;
  if (idx >= total) return;
  long sb = (long)blockIdx.z * total;
  int k = (int)(idx / N);
  int n = (int)(idx % N);
  Wt[sb + (long)n * K + k] = f2bf(W[sb + idx]);
}

// ---------------------------------------------------------------------------
// State init.
// ---------------------------------------------------------------------------
__global__ void k_init(const float* __restrict__ deter0,
                       const float* __restrict__ stoch0,
                       float* __restrict__ deter, bf16_t* __restrict__ catbuf,
                       bf16_t* __restrict__ stoch) {
  int idx = blockIdx.x * 256 + threadIdx.x;
  if (idx < BDIM * DDIM) {
    float v = deter0[idx];
    deter[idx] = v;
    int row = idx / DDIM, j = idx % DDIM;
    catbuf[(long)row * (HDIM + DDIM) + HDIM + j] = f2bf(v);
  }
  if (idx < BDIM * SDIM) stoch[idx] = f2bf(stoch0[idx]);
}

// ---------------------------------------------------------------------------
// Build xin = [stoch(bf16) | act_t(->bf16)]  : [B, S+A]
// ---------------------------------------------------------------------------
__global__ void k_prep(const bf16_t* __restrict__ stoch,
                       const float* __restrict__ act_t,
                       bf16_t* __restrict__ xin) {
  int idx = blockIdx.x * 256 + threadIdx.x;
  if (idx >= BDIM * (SDIM + ADIM)) return;
  int row = idx / (SDIM + ADIM);
  int col = idx % (SDIM + ADIM);
  xin[idx] = (col < SDIM) ? stoch[row * SDIM + col]
                          : f2bf(act_t[row * ADIM + (col - SDIM)]);
}

// ---------------------------------------------------------------------------
// Register-tiled WMMA GEMM (direct global loads).
// C[M,N](f32) = A[M,K](bf16,row-major) x Bt[N,K](bf16 row-major = B^T).
// Wave: 32x32 tile (2 A-frags x 2 B-frags -> 4 WMMAs / 32-K step).
// Block: 256 thr = 8 waves as 2(M) x 4(N) -> 64 x 128 tile.
// ---------------------------------------------------------------------------
__global__ void k_gemm_bf16(const bf16_t* __restrict__ Abase, int lda,
                            const bf16_t* __restrict__ Btbase, int ldbt,
                            float* __restrict__ C, int ldc,
                            int M, int N, int K,
                            const int* __restrict__ ens_idx, int t,
                            long wslice) {
  const bf16_t* Bt = Btbase;
  if (ens_idx) Bt += (long)ens_idx[t] * wslice;

  int lane = threadIdx.x & 31;
  int wave = threadIdx.x >> 5;
  int wm = wave >> 2; // 0..1
  int wn = wave & 3;  // 0..3
  int row0 = blockIdx.y * 64 + wm * 32;
  int n0 = blockIdx.x * 128 + wn * 32;

  int arow = lane & 15;
  int ahi = (lane >> 4) * 8;
  const bf16_t* pA0 = Abase + (long)(row0 + arow) * lda + ahi;
  const bf16_t* pA1 = pA0 + (long)16 * lda;

  int bcol = lane & 15;
  int bkhi = (lane >> 4) * 8;
  const bf16_t* pB0 = Bt + (long)(n0 + bcol) * ldbt + bkhi;
  const bf16_t* pB1 = pB0 + (long)16 * ldbt;

  v8f acc00 = {0.f, 0.f, 0.f, 0.f, 0.f, 0.f, 0.f, 0.f};
  v8f acc01 = acc00, acc10 = acc00, acc11 = acc00;

  for (int k0 = 0; k0 < K; k0 += 32) {
    v16bf a0 = load_afrag(pA0);
    v16bf a1 = load_afrag(pA1);
    v16bf b0 = load_afrag(pB0);
    v16bf b1 = load_afrag(pB1);
    __builtin_prefetch((const void*)(pB0 + 512), 0, 0); // global_prefetch_b8
    acc00 = wmma_bf16(a0, b0, acc00);
    acc01 = wmma_bf16(a0, b1, acc01);
    acc10 = wmma_bf16(a1, b0, acc10);
    acc11 = wmma_bf16(a1, b1, acc11);
    pA0 += 32; pA1 += 32; pB0 += 32; pB1 += 32;
  }
  store_cfrag(C, ldc, row0, n0, lane, acc00);
  store_cfrag(C, ldc, row0, n0 + 16, lane, acc01);
  store_cfrag(C, ldc, row0 + 16, n0, lane, acc10);
  store_cfrag(C, ldc, row0 + 16, n0 + 16, lane, acc11);
}

// ---------------------------------------------------------------------------
// Same tiling, but the 64x32 A tile is staged in LDS with
// global_load_async_to_lds_b128 (double-buffered, ASYNCcnt-tracked); each of
// 256 threads moves 16B per K-step. LDS rows padded to 40 bf16 so the
// ds_load_b128 fragment reads are bank-conflict free. B streams directly
// from the pre-transposed bf16 weights (contiguous 32B loads, L2-resident).
// ---------------------------------------------------------------------------
__global__ void k_gemm_bf16_lds(const bf16_t* __restrict__ Abase, int lda,
                                const bf16_t* __restrict__ Btbase, int ldbt,
                                float* __restrict__ C, int ldc,
                                int M, int N, int K,
                                const int* __restrict__ ens_idx, int t,
                                long wslice) {
  __shared__ __align__(16) bf16_t As[2][64 * LDS_AROW];

  const bf16_t* Bt = Btbase;
  if (ens_idx) Bt += (long)ens_idx[t] * wslice;

  int tid = threadIdx.x;
  int lane = tid & 31;
  int wave = tid >> 5;
  int wm = wave >> 2;
  int wn = wave & 3;
  int blockRow = blockIdx.y * 64;
  int n0 = blockIdx.x * 128 + wn * 32;

  // Cooperative A staging: thread -> (row = tid/4, 8-elem group = tid%4).
  int ldrow = tid >> 2;       // 0..63
  int ldcol = (tid & 3) * 8;  // 0,8,16,24
  const bf16_t* gA = Abase + (long)(blockRow + ldrow) * lda + ldcol;
  // Generic LDS pointers: addr[31:0] is the LDS byte offset (flat aperture).
  unsigned lds0 = (unsigned)(size_t)(void*)&As[0][ldrow * LDS_AROW + ldcol];
  unsigned lds1 = (unsigned)(size_t)(void*)&As[1][ldrow * LDS_AROW + ldcol];

#define ASYNC_A(LDS_OFF, GPTR)                                              \
  asm volatile("global_load_async_to_lds_b128 %0, %1, off"                  \
               :: "v"(LDS_OFF), "v"((unsigned long)(size_t)(GPTR))          \
               : "memory")

  ASYNC_A(lds0, gA);

  int bcol = lane & 15;
  int bkhi = (lane >> 4) * 8;
  const bf16_t* pB0 = Bt + (long)(n0 + bcol) * ldbt + bkhi;
  const bf16_t* pB1 = pB0 + (long)16 * ldbt;

  int arowl = wm * 32 + (lane & 15); // local row of this wave's A frags
  int ahi = (lane >> 4) * 8;

  v8f acc00 = {0.f, 0.f, 0.f, 0.f, 0.f, 0.f, 0.f, 0.f};
  v8f acc01 = acc00, acc10 = acc00, acc11 = acc00;

  int nk = K / 32;
  for (int ki = 0; ki < nk; ++ki) {
    int buf = ki & 1;
    if (ki + 1 < nk) {
      // Prefetch next A tile into the other buffer, then wait for current.
      ASYNC_A(buf ? lds0 : lds1, gA + (long)(ki + 1) * 32);
      asm volatile("s_wait_asynccnt 0x1" ::: "memory");
    } else {
      asm volatile("s_wait_asynccnt 0x0" ::: "memory");
    }
    __syncthreads(); // all waves' portions of As[buf] visible

    const bf16_t* sA0 = &As[buf][arowl * LDS_AROW + ahi];
    const bf16_t* sA1 = sA0 + 16 * LDS_AROW;
    v16bf a0 = load_afrag(sA0); // ds_load_b128 pair
    v16bf a1 = load_afrag(sA1);
    v16bf b0 = load_afrag(pB0);
    v16bf b1 = load_afrag(pB1);
    __builtin_prefetch((const void*)(pB0 + 512), 0, 0);
    acc00 = wmma_bf16(a0, b0, acc00);
    acc01 = wmma_bf16(a0, b1, acc01);
    acc10 = wmma_bf16(a1, b0, acc10);
    acc11 = wmma_bf16(a1, b1, acc11);
    pB0 += 32; pB1 += 32;
    __syncthreads(); // As[buf] fully consumed before next overwrite issue
  }
#undef ASYNC_A

  int row0 = blockRow + wm * 32;
  store_cfrag(C, ldc, row0, n0, lane, acc00);
  store_cfrag(C, ldc, row0, n0 + 16, lane, acc01);
  store_cfrag(C, ldc, row0 + 16, n0, lane, acc10);
  store_cfrag(C, ldc, row0 + 16, n0 + 16, lane, acc11);
}

// ---------------------------------------------------------------------------
// Row LayerNorm (pre-LN bias) + ELU -> bf16 (strided dst); optional ensemble
// parameter indexing. One block (256 thr) per row.
// ---------------------------------------------------------------------------
__global__ void k_ln_act(const float* __restrict__ X, int ldx,
                         const float* __restrict__ b,
                         const float* __restrict__ g,
                         const float* __restrict__ be,
                         bf16_t* __restrict__ dst, int ldd, int N,
                         const int* __restrict__ ens_idx, int t, int pslice) {
  int row = blockIdx.x, tid = threadIdx.x;
  if (ens_idx) {
    long o = (long)ens_idx[t] * pslice;
    b += o; g += o; be += o;
  }
  __shared__ float s1[256], s2[256];
  const float* Xr = X + (long)row * ldx;
  float sum = 0.f, sq = 0.f;
  for (int j = tid; j < N; j += 256) {
    float v = Xr[j] + b[j];
    sum += v; sq += v * v;
  }
  s1[tid] = sum; s2[tid] = sq;
  __syncthreads();
  for (int s = 128; s > 0; s >>= 1) {
    if (tid < s) { s1[tid] += s1[tid + s]; s2[tid] += s2[tid + s]; }
    __syncthreads();
  }
  float mu = s1[0] / N;
  float rstd = rsqrtf(s2[0] / N - mu * mu + 1e-5f);
  for (int j = tid; j < N; j += 256) {
    float v = Xr[j] + b[j];
    float y = (v - mu) * rstd * g[j] + be[j];
    y = (y > 0.f) ? y : expm1f(y); // ELU
    dst[(long)row * ldd + j] = f2bf(y);
  }
}

// ---------------------------------------------------------------------------
// GRU cell epilogue: LN(parts+b) over 3D, split r/c/u, update deter.
// ---------------------------------------------------------------------------
__global__ void k_gru(const float* __restrict__ parts,
                      const float* __restrict__ b, const float* __restrict__ g,
                      const float* __restrict__ be,
                      float* __restrict__ deter,
                      const float* __restrict__ obs_t,
                      float* __restrict__ out_t,
                      bf16_t* __restrict__ catbuf,
                      bf16_t* __restrict__ postcat) {
  int row = blockIdx.x, tid = threadIdx.x;
  const int N3 = 3 * DDIM;
  __shared__ float s1[256], s2[256];
  const float* X = parts + (long)row * N3;
  float sum = 0.f, sq = 0.f;
  for (int j = tid; j < N3; j += 256) {
    float v = X[j] + b[j];
    sum += v; sq += v * v;
  }
  s1[tid] = sum; s2[tid] = sq;
  __syncthreads();
  for (int s = 128; s > 0; s >>= 1) {
    if (tid < s) { s1[tid] += s1[tid + s]; s2[tid] += s2[tid + s]; }
    __syncthreads();
  }
  float mu = s1[0] / N3;
  float rstd = rsqrtf(s2[0] / N3 - mu * mu + 1e-5f);
  for (int j = tid; j < DDIM; j += 256) {
    float rr = ((X[j] + b[j]) - mu) * rstd * g[j] + be[j];
    float cc = ((X[j + DDIM] + b[j + DDIM]) - mu) * rstd * g[j + DDIM] + be[j + DDIM];
    float uu = ((X[j + 2 * DDIM] + b[j + 2 * DDIM]) - mu) * rstd * g[j + 2 * DDIM] + be[j + 2 * DDIM];
    float reset = 1.f / (1.f + expf(-rr));
    float cand = tanhf(reset * cc);
    float update = 1.f / (1.f + expf(-(uu - 1.f)));
    float dold = deter[(long)row * DDIM + j];
    float dnew = update * cand + (1.f - update) * dold;
    deter[(long)row * DDIM + j] = dnew;
    out_t[(long)row * OUTC + 4 * SDIM + j] = dnew;
    bf16_t dbf = f2bf(dnew);
    catbuf[(long)row * (HDIM + DDIM) + HDIM + j] = dbf;
    postcat[(long)row * (DDIM + ODIM) + j] = dbf;
  }
  for (int j = tid; j < ODIM; j += 256) {
    postcat[(long)row * (DDIM + ODIM) + DDIM + j] = f2bf(obs_t[(long)row * ODIM + j]);
  }
}

// ---------------------------------------------------------------------------
// Head epilogue: mean = X[:,:S]+b; std = softplus(X[:,S:]+b)+0.1; optional
// rsample stoch = mean + std*noise (bf16 state for next step).
// ---------------------------------------------------------------------------
__global__ void k_head2(const float* __restrict__ X,
                        const float* __restrict__ bias,
                        const int* __restrict__ ens_idx, int t, int bslice,
                        float* __restrict__ out_t, int off_mean, int off_std,
                        const float* __restrict__ noise_t,
                        bf16_t* __restrict__ stoch) {
  int idx = blockIdx.x * 256 + threadIdx.x;
  if (idx >= BDIM * SDIM) return;
  int row = idx >> 6;
  int jj = idx & 63;
  const float* bb = bias;
  if (ens_idx) bb += (long)ens_idx[t] * bslice;
  float m = X[row * 2 * SDIM + jj] + bb[jj];
  float sraw = X[row * 2 * SDIM + SDIM + jj] + bb[SDIM + jj];
  float sp = (sraw > 20.f) ? sraw : log1pf(expf(sraw)); // softplus
  float s = sp + 0.1f;
  out_t[(long)row * OUTC + off_mean + jj] = m;
  out_t[(long)row * OUTC + off_std + jj] = s;
  if (noise_t) stoch[row * SDIM + jj] = f2bf(m + s * noise_t[row * SDIM + jj]);
}

// ---------------------------------------------------------------------------
// Host orchestration
// ---------------------------------------------------------------------------
extern "C" void kernel_launch(void* const* d_in, const int* in_sizes, int n_in,
                              void* d_out, int out_size, void* d_ws,
                              size_t ws_size, hipStream_t stream) {
  (void)in_sizes; (void)n_in; (void)out_size; (void)ws_size;
  const float* obs    = (const float*)d_in[0];
  const float* act    = (const float*)d_in[1];
  const float* deter0 = (const float*)d_in[2];
  const float* stoch0 = (const float*)d_in[3];
  const float* noise  = (const float*)d_in[4];
  const float* W_img  = (const float*)d_in[5];
  const float* b_img  = (const float*)d_in[6];
  const float* g_img  = (const float*)d_in[7];
  const float* be_img = (const float*)d_in[8];
  const float* W_gru  = (const float*)d_in[9];
  const float* b_gru  = (const float*)d_in[10];
  const float* g_gru  = (const float*)d_in[11];
  const float* be_gru = (const float*)d_in[12];
  const float* W_eo1  = (const float*)d_in[13];
  const float* b_eo1  = (const float*)d_in[14];
  const float* g_eo1  = (const float*)d_in[15];
  const float* be_eo1 = (const float*)d_in[16];
  const float* W_eo2  = (const float*)d_in[17];
  const float* b_eo2  = (const float*)d_in[18];
  const float* W_oo1  = (const float*)d_in[19];
  const float* b_oo1  = (const float*)d_in[20];
  const float* g_oo1  = (const float*)d_in[21];
  const float* be_oo1 = (const float*)d_in[22];
  const float* W_oo2  = (const float*)d_in[23];
  const float* b_oo2  = (const float*)d_in[24];
  const int*   ens    = (const int*)d_in[25];
  float* out = (float*)d_out;

  char* wsp = (char*)d_ws;
  size_t off = 0;
  auto alloc = [&](size_t bytes) -> void* {
    off = (off + 255) & ~(size_t)255;
    void* p = wsp + off;
    off += bytes;
    return p;
  };
  bf16_t* Wt_img = (bf16_t*)alloc((size_t)HDIM * (SDIM + ADIM) * 2);
  bf16_t* Wt_gru = (bf16_t*)alloc((size_t)3 * DDIM * (HDIM + DDIM) * 2);
  bf16_t* Wt_eo1 = (bf16_t*)alloc((size_t)EDIM * HDIM * DDIM * 2);
  bf16_t* Wt_eo2 = (bf16_t*)alloc((size_t)EDIM * 2 * SDIM * HDIM * 2);
  bf16_t* Wt_oo1 = (bf16_t*)alloc((size_t)HDIM * (DDIM + ODIM) * 2);
  bf16_t* Wt_oo2 = (bf16_t*)alloc((size_t)2 * SDIM * HDIM * 2);
  bf16_t* xin     = (bf16_t*)alloc((size_t)BDIM * (SDIM + ADIM) * 2);
  bf16_t* catbuf  = (bf16_t*)alloc((size_t)BDIM * (HDIM + DDIM) * 2);
  bf16_t* postcat = (bf16_t*)alloc((size_t)BDIM * (DDIM + ODIM) * 2);
  bf16_t* hbuf    = (bf16_t*)alloc((size_t)BDIM * HDIM * 2);
  bf16_t* stoch   = (bf16_t*)alloc((size_t)BDIM * SDIM * 2);
  float*  deter   = (float*)alloc((size_t)BDIM * DDIM * 4);
  float*  tmp1    = (float*)alloc((size_t)BDIM * HDIM * 4);
  float*  tmp2    = (float*)alloc((size_t)BDIM * 3 * DDIM * 4);
  float*  tmp3    = (float*)alloc((size_t)BDIM * 2 * SDIM * 4);

  auto w2bft = [&](const float* W, bf16_t* Wt, int K, int N, int batch) {
    long total = (long)K * N;
    dim3 grid((unsigned)((total + 255) / 256), 1, (unsigned)batch);
    k_w2bft<<<grid, dim3(256), 0, stream>>>(W, Wt, K, N);
  };
  w2bft(W_img, Wt_img, SDIM + ADIM, HDIM, 1);
  w2bft(W_gru, Wt_gru, HDIM + DDIM, 3 * DDIM, 1);
  w2bft(W_eo1, Wt_eo1, DDIM, HDIM, EDIM);
  w2bft(W_eo2, Wt_eo2, HDIM, 2 * SDIM, EDIM);
  w2bft(W_oo1, Wt_oo1, DDIM + ODIM, HDIM, 1);
  w2bft(W_oo2, Wt_oo2, HDIM, 2 * SDIM, 1);

  k_init<<<dim3((BDIM * DDIM + 255) / 256), dim3(256), 0, stream>>>(
      deter0, stoch0, deter, catbuf, stoch);

  // Direct-load GEMM (small K / tiny heads).
  auto gemm = [&](const bf16_t* Ap, int lda, const bf16_t* Btp, int ldbt,
                  float* Cp, int ldc, int M, int N, int K, const int* ei,
                  int t, long wsl) {
    dim3 grid((unsigned)(N / 128), (unsigned)(M / 64));
    k_gemm_bf16<<<grid, dim3(256), 0, stream>>>(Ap, lda, Btp, ldbt, Cp, ldc,
                                                M, N, K, ei, t, wsl);
  };
  // LDS-staged GEMM (big K, async A tile).
  auto gemm_lds = [&](const bf16_t* Ap, int lda, const bf16_t* Btp, int ldbt,
                      float* Cp, int ldc, int M, int N, int K, const int* ei,
                      int t, long wsl) {
    dim3 grid((unsigned)(N / 128), (unsigned)(M / 64));
    k_gemm_bf16_lds<<<grid, dim3(256), 0, stream>>>(Ap, lda, Btp, ldbt, Cp,
                                                    ldc, M, N, K, ei, t, wsl);
  };

  for (int t = 0; t < TDIM; ++t) {
    const float* act_t   = act + (size_t)t * BDIM * ADIM;
    const float* obs_t   = obs + (size_t)t * BDIM * ODIM;
    const float* noise_t = noise + (size_t)t * BDIM * SDIM;
    float* out_t = out + (size_t)t * BDIM * OUTC;

    // _img_in: ELU(LN(cat(stoch,act) @ W + b)) -> catbuf[:, :H] (bf16)
    k_prep<<<dim3((BDIM * (SDIM + ADIM) + 255) / 256), dim3(256), 0, stream>>>(
        stoch, act_t, xin);
    gemm(xin, SDIM + ADIM, Wt_img, SDIM + ADIM, tmp1, HDIM, BDIM, HDIM,
         SDIM + ADIM, nullptr, 0, 0);
    k_ln_act<<<dim3(BDIM), dim3(256), 0, stream>>>(
        tmp1, HDIM, b_img, g_img, be_img, catbuf, HDIM + DDIM, HDIM,
        (const int*)nullptr, 0, 0);

    // norm-GRU: LN(cat(x,deter) @ W_gru + b) -> r,c,u -> new deter
    gemm_lds(catbuf, HDIM + DDIM, Wt_gru, HDIM + DDIM, tmp2, 3 * DDIM, BDIM,
             3 * DDIM, HDIM + DDIM, nullptr, 0, 0);
    k_gru<<<dim3(BDIM), dim3(256), 0, stream>>>(
        tmp2, b_gru, g_gru, be_gru, deter, obs_t, out_t, catbuf, postcat);

    // prior head (ensemble member ens_idx[t], resolved on-device)
    gemm_lds(catbuf + HDIM, HDIM + DDIM, Wt_eo1, DDIM, tmp1, HDIM, BDIM, HDIM,
             DDIM, ens, t, (long)HDIM * DDIM);
    k_ln_act<<<dim3(BDIM), dim3(256), 0, stream>>>(
        tmp1, HDIM, b_eo1, g_eo1, be_eo1, hbuf, HDIM, HDIM, ens, t, HDIM);
    gemm(hbuf, HDIM, Wt_eo2, HDIM, tmp3, 2 * SDIM, BDIM, 2 * SDIM, HDIM, ens,
         t, (long)2 * SDIM * HDIM);
    k_head2<<<dim3((BDIM * SDIM + 255) / 256), dim3(256), 0, stream>>>(
        tmp3, b_eo2, ens, t, 2 * SDIM, out_t, 2 * SDIM, 3 * SDIM,
        (const float*)nullptr, (bf16_t*)nullptr);

    // posterior head: cat(deter, obs) -> qm, post_std; rsample -> stoch
    gemm_lds(postcat, DDIM + ODIM, Wt_oo1, DDIM + ODIM, tmp1, HDIM, BDIM,
             HDIM, DDIM + ODIM, nullptr, 0, 0);
    k_ln_act<<<dim3(BDIM), dim3(256), 0, stream>>>(
        tmp1, HDIM, b_oo1, g_oo1, be_oo1, hbuf, HDIM, HDIM,
        (const int*)nullptr, 0, 0);
    gemm(hbuf, HDIM, Wt_oo2, HDIM, tmp3, 2 * SDIM, BDIM, 2 * SDIM, HDIM,
         nullptr, 0, 0);
    k_head2<<<dim3((BDIM * SDIM + 255) / 256), dim3(256), 0, stream>>>(
        tmp3, b_oo2, (const int*)nullptr, 0, 0, out_t, 0, SDIM, noise_t,
        stoch);
  }
}
